// FlowEquivariantMambaCell_54820962566295
// MI455X (gfx1250) — matrix operand order repaired and checked by reading, not compiled, not run on hardware
//
#include <hip/hip_runtime.h>
#include <hip/hip_bf16.h>
#include <math.h>
#include <stdint.h>

typedef __attribute__((ext_vector_type(16))) __bf16 v16bf;
typedef __attribute__((ext_vector_type(8)))  float  v8f;

#define D_MODEL 64
#define D_STATE 16
#define NVEL    9
#define HW      1024
#define BATCH   4
#define KTOT    576      // 9 taps * 64 in-channels
#define LDSW_STRIDE 17   // pad 16 -> 17 dwords per k-row (bank de-alias)

__device__ __forceinline__ unsigned short bf16_bits(__bf16 x) {
    unsigned short s; __builtin_memcpy(&s, &x, 2); return s;
}
__device__ __forceinline__ __bf16 bits_bf16(unsigned short s) {
    __bf16 x; __builtin_memcpy(&x, &s, 2); return x;
}
__device__ __forceinline__ uint32_t split_pack(float v) {
    __bf16 hi = (__bf16)v;
    __bf16 lo = (__bf16)(v - (float)hi);
    return ((uint32_t)bf16_bits(lo) << 16) | (uint32_t)bf16_bits(hi);
}

// ---------------------------------------------------------------------------
// Phase 0: pre-split u into packed (hi,lo) bf16 pairs (1 MB, done once).
// ---------------------------------------------------------------------------
__global__ __launch_bounds__(256)
void fe_mamba_pack_u(const float* __restrict__ u, uint32_t* __restrict__ upack)
{
    const int i = blockIdx.x * blockDim.x + threadIdx.x;   // 262144
    upack[i] = split_pack(u[i]);
}

// ---------------------------------------------------------------------------
// Phase 1: fused circular 3x3 convs (delta / B / C) as implicit GEMM on WMMA.
//   Y[96 x 4096] = Wcat[96 x 576] * X[576 x 4096],  k = kk*64 + ic
// Block = 1 M-tile (16 out-channels) x 4 N-tiles (one per wave).
// Weights staged once per block into LDS as pre-split packed bf16 (hi,lo).
// bf16x3 compensated products ~= fp32 accuracy on v_wmma_f32_16x16x32_bf16.
// Fragment K-maps per CDNA5 ISA 7.12.2:
//   A (16x32, 16-bit): lane<16 -> K = (e<8 ? e : e+8); lane>=16 -> +8
//   B (32x16, 16-bit): lane<16 -> K = (e<8 ? e : e+8); lane>=16 -> +8
// ---------------------------------------------------------------------------
__global__ __launch_bounds__(128)
void fe_mamba_conv_wmma(const uint32_t* __restrict__ upack,
                        const float* __restrict__ Wd,
                        const float* __restrict__ bd,
                        const float* __restrict__ Wb,
                        const float* __restrict__ Wc,
                        const float* __restrict__ dt_bias,
                        float* __restrict__ delta_ws,
                        float* __restrict__ Bval_ws,
                        float* __restrict__ Cval_ws)
{
    __shared__ uint32_t lds_w[KTOT * LDSW_STRIDE];   // [k][m], packed (hi,lo)

    const int mtile  = blockIdx.x % 6;
    const int nblock = blockIdx.x / 6;               // 64 blocks of 4 ntiles

    // ---- stage 16x576 weight slab for this M-tile into LDS (split once) ----
    for (int j = threadIdx.x; j < 16 * KTOT; j += 128) {
        const int m  = j / KTOT;
        const int k  = j % KTOT;
        const int kk = k >> 6;
        const int ic = k & 63;
        const int oc = mtile * 16 + m;
        float wv;
        if (oc < 64)      wv = Wd[(oc * 64 + ic) * 9 + kk];
        else if (oc < 80) wv = Wb[((oc - 64) * 64 + ic) * 9 + kk];
        else              wv = Wc[((oc - 80) * 64 + ic) * 9 + kk];
        lds_w[k * LDSW_STRIDE + m] = split_pack(wv);
    }
    __syncthreads();

    const int lane   = threadIdx.x & 31;
    const int waveid = threadIdx.x >> 5;
    const int ntile  = nblock * 4 + waveid;
    const int hi16   = lane >> 4;

    // B-fragment lane -> spatial position
    const int ncol = lane & 15;
    const int p    = ntile * 16 + ncol;
    const int b    = p >> 10;
    const int h    = (p >> 5) & 31;
    const int w    = p & 31;
    // A-fragment lane -> output-channel row within M-tile
    const int m = lane & 15;

    v8f acc = {};

    #pragma unroll
    for (int kk = 0; kk < 9; ++kk) {
        const int ky = kk / 3, kx = kk % 3;
        const int hs = (h + ky + 31) & 31;           // circular pad
        const int ws = (w + kx + 31) & 31;
        const uint32_t* usrc = upack + (size_t)b * D_MODEL * HW + hs * 32 + ws;
        #pragma unroll
        for (int half = 0; half < 2; ++half) {
            const int c0 = half * 32;
            v16bf a_hi, a_lo, b_hi, b_lo;
            #pragma unroll
            for (int e = 0; e < 16; ++e) {
                const int ka = (e < 8 ? e : e + 8) + (hi16 << 3);
                const uint32_t aw = lds_w[(kk * 64 + c0 + ka) * LDSW_STRIDE + m];
                a_hi[e] = bits_bf16((unsigned short)(aw & 0xffffu));
                a_lo[e] = bits_bf16((unsigned short)(aw >> 16));
                const int kb = (e < 8 ? e : e + 8) + (hi16 << 3);
                const uint32_t xw = usrc[(c0 + kb) * HW];
                b_hi[e] = bits_bf16((unsigned short)(xw & 0xffffu));
                b_lo[e] = bits_bf16((unsigned short)(xw >> 16));
            }
            acc = __builtin_amdgcn_wmma_f32_16x16x32_bf16(false, a_lo, false, b_hi,
                                                          (short)0, acc, false, false);
            acc = __builtin_amdgcn_wmma_f32_16x16x32_bf16(false, a_hi, false, b_lo,
                                                          (short)0, acc, false, false);
            acc = __builtin_amdgcn_wmma_f32_16x16x32_bf16(false, a_hi, false, b_hi,
                                                          (short)0, acc, false, false);
        }
    }

    // Epilogue: C/D layout -> lane holds N=lane&15, VGPR r holds M = r + 8*hi16
    const float dtb = dt_bias[0];
    const int hw = h * 32 + w;
    #pragma unroll
    for (int r = 0; r < 8; ++r) {
        const int mrow = r + (hi16 << 3);
        const int ocr  = mtile * 16 + mrow;
        const float val = acc[r];
        if (ocr < 64) {
            float x  = val + bd[ocr] + dtb;
            float sp = (x > 20.f) ? x : log1pf(expf(x));     // stable softplus
            delta_ws[(b * 64 + ocr) * HW + hw] = sp;
        } else if (ocr < 80) {
            Bval_ws[(b * 16 + (ocr - 64)) * HW + hw] = val;
        } else {
            Cval_ws[(b * 16 + (ocr - 80)) * HW + hw] = val;
        }
    }
}

// ---------------------------------------------------------------------------
// Phase 2: bandwidth-bound state stream (302 MB touched once -> NT hints).
// One thread per (b,d,h,w): s_next = A_bar * roll(s_prev) + drive;
// y = max_v( sum_n s_next*C + u*D ).
// ---------------------------------------------------------------------------
__global__ __launch_bounds__(256)
void fe_mamba_scan(const float* __restrict__ u,
                   const float* __restrict__ s_prev,
                   const float* __restrict__ Dp,
                   const float* __restrict__ logA,
                   const float* __restrict__ delta_ws,
                   const float* __restrict__ Bval_ws,
                   const float* __restrict__ Cval_ws,
                   float* __restrict__ y_out,
                   float* __restrict__ s_next)
{
    const int idx = blockIdx.x * blockDim.x + threadIdx.x;   // B*64*HW = 262144
    const int w = idx & 31;
    const int h = (idx >> 5) & 31;
    const int d = (idx >> 10) & 63;
    const int b = idx >> 16;
    const int hw = h * 32 + w;

    const float uv    = u[(b * 64 + d) * HW + hw];
    const float delta = delta_ws[(b * 64 + d) * HW + hw];

    float Abar[D_STATE], drive[D_STATE], Cv[D_STATE];
    #pragma unroll
    for (int n = 0; n < D_STATE; ++n) {
        const float Aval = -expf(logA[d * D_STATE + n]);   // A = -exp(log_A_real)
        const float ab   = expf(delta * Aval);             // A_bar
        Abar[n] = ab;
        const float bv = Bval_ws[(b * 16 + n) * HW + hw];
        drive[n] = (ab - 1.0f) / Aval * bv * uv;           // B_bar * u
        Cv[n]    = Cval_ws[(b * 16 + n) * HW + hw];
    }

    const float yD = uv * Dp[d];
    float ymax = -__builtin_inff();

    for (int v = 0; v < NVEL; ++v) {
        const int vx = v / 3 - 1;                          // V_LIST is x-major
        const int vy = v % 3 - 1;
        const int hs = (h - vy + 32) & 31;                 // roll(s,(vy,vx)) gather
        const int ws = (w - vx + 32) & 31;
        const size_t base = ((size_t)(b * NVEL + v) * D_MODEL + d) * D_STATE * HW;
        const float* sp = s_prev + base + hs * 32 + ws;
        float*       sn = s_next + base + hw;
        if (v + 1 < NVEL) {                                // hide next-v latency
            const size_t nbase = ((size_t)(b * NVEL + v + 1) * D_MODEL + d) * D_STATE * HW;
            __builtin_prefetch(s_prev + nbase + hs * 32 + ws, 0, 0);
        }
        float yv = 0.f;
        #pragma unroll
        for (int n = 0; n < D_STATE; ++n) {
            const float s  = __builtin_nontemporal_load(sp + n * HW);
            const float ns = fmaf(Abar[n], s, drive[n]);
            __builtin_nontemporal_store(ns, sn + n * HW);
            yv = fmaf(ns, Cv[n], yv);
        }
        ymax = fmaxf(ymax, yv + yD);
    }
    y_out[(b * 64 + d) * HW + hw] = ymax;
}

extern "C" void kernel_launch(void* const* d_in, const int* in_sizes, int n_in,
                              void* d_out, int out_size, void* d_ws, size_t ws_size,
                              hipStream_t stream) {
    (void)in_sizes; (void)n_in; (void)out_size; (void)ws_size;
    const float* u      = (const float*)d_in[0];
    const float* s_prev = (const float*)d_in[1];
    const float* Wd     = (const float*)d_in[2];
    const float* bd     = (const float*)d_in[3];
    const float* Wb     = (const float*)d_in[4];
    const float* Wc     = (const float*)d_in[5];
    const float* Dp     = (const float*)d_in[6];
    const float* logA   = (const float*)d_in[7];
    const float* dtb    = (const float*)d_in[8];

    float*    delta_ws = (float*)d_ws;                        // 262144 f32
    float*    Bval_ws  = delta_ws + BATCH * D_MODEL * HW;     //  65536 f32
    float*    Cval_ws  = Bval_ws + BATCH * D_STATE * HW;      //  65536 f32
    uint32_t* upack    = (uint32_t*)(Cval_ws + BATCH * D_STATE * HW); // 262144 u32

    float* y_out  = (float*)d_out;                            // (4,64,32,32)
    float* s_next = y_out + BATCH * D_MODEL * HW;             // (4,9,64,16,32,32)

    fe_mamba_pack_u<<<1024, 256, 0, stream>>>(u, upack);
    // 6 M-tiles x 64 N-blocks (4 waves = 4 N-tiles each) = 384 blocks.
    fe_mamba_conv_wmma<<<384, 128, 0, stream>>>(upack, Wd, bd, Wb, Wc, dtb,
                                                delta_ws, Bval_ws, Cval_ws);
    fe_mamba_scan<<<1024, 256, 0, stream>>>(u, s_prev, Dp, logA,
                                            delta_ws, Bval_ws, Cval_ws,
                                            y_out, s_next);
}
